// ContinuousTimeRNN_9474697855223
// MI455X (gfx1250) — compile-verified, matrix operand order checked by reading, not run on hardware
//
#include <hip/hip_runtime.h>
#include <hip/hip_bf16.h>

// Problem constants (from reference)
#define B_ 256
#define S_ 128
#define I_ 256
#define H_ 2048
#define O_ 256
#define KCHUNK 512   // K-tile staged in LDS: 32 rows * 512 * 2B = 32 KB

typedef __attribute__((ext_vector_type(16))) __bf16 v16bf;
typedef __attribute__((ext_vector_type(8)))  __bf16 v8bf;
typedef __attribute__((ext_vector_type(8)))  float  v8f;

__device__ __forceinline__ v16bf cat16(v8bf lo, v8bf hi) {
  return __builtin_shufflevector(lo, hi, 0,1,2,3,4,5,6,7,8,9,10,11,12,13,14,15);
}

// ---------------------------------------------------------------------------
// Fused epilogue. MODE: 0 = input GEMM (out = acc + b_in)
//                        1..4 = RK4 evals (k = tanh(acc+inp) - heval; update state)
//                        5 = decode GEMM (out = acc + b_dec)
// acc element r corresponds to state row (mrow0 + r), column `col`.
// ---------------------------------------------------------------------------
template<int MODE>
__device__ __forceinline__ void epilogue(
    const v8f acc, unsigned mrow0, unsigned col,
    const float* __restrict__ bias, const float* __restrict__ inp,
    float* __restrict__ h, float* __restrict__ heval, float* __restrict__ ksum,
    __bf16* __restrict__ hb_out, float* __restrict__ out, unsigned ldout)
{
#pragma unroll
  for (int r = 0; r < 8; ++r) {
    const unsigned row = mrow0 + r;
    const float a = acc[r];
    if (MODE == 0 || MODE == 5) {
      out[row * ldout + col] = a + bias[col];
    } else {
      const unsigned o = row * (unsigned)H_ + col;
      const float hv = h[o];
      const float he = heval[o];
      const float k  = tanhf(a + inp[o]) - he;    // f(h_eval, inp), tau=1
      float hn;
      if (MODE == 1)      { ksum[o] = k;                 hn = hv + 0.5f * k; }
      else if (MODE == 2) { ksum[o] += 2.0f * k;         hn = hv + 0.5f * k; }
      else if (MODE == 3) { ksum[o] += 2.0f * k;         hn = hv + k;        }
      else /* MODE==4 */  { const float ks = ksum[o] + k;
                            hn = hv + ks * (1.0f / 6.0f);
                            h[o] = hn; }
      heval[o]  = hn;              // h_eval for next RK4 stage (or h for next step)
      hb_out[o] = (__bf16)hn;      // bf16 A-operand for next GEMM (double-buffered)
    }
  }
}

// ---------------------------------------------------------------------------
// Unified bf16 WMMA GEMM:  C(256 x N) = A(256 x K, bf16 row-major)
//                                       @ W(N x K, bf16 row-major)^T
// grid.x = N/32 (one 32-wide N stripe per block), block = 256 threads = 8 waves.
// Each wave computes a 2x2 block of 16x16 tiles; each A/B fragment feeds two
// WMMAs: 4 v_wmma per 8 fragment loads per k-step.
// W stripe (32 rows x KCHUNK) is staged into LDS with the CDNA5 async
// direct-to-LDS path (GLOBAL_LOAD_ASYNC_TO_LDS_B128, ASYNCcnt) — no VGPR
// round-trip for weight data.
// ---------------------------------------------------------------------------
template<int MODE>
__global__ __launch_bounds__(256)
void rk4_gemm(const __bf16* __restrict__ A, const __bf16* __restrict__ W, int K,
              const float* __restrict__ bias, const float* __restrict__ inp,
              float* __restrict__ h, float* __restrict__ heval,
              float* __restrict__ ksum, __bf16* __restrict__ hb_out,
              float* __restrict__ out, unsigned ldout)
{
  __shared__ __align__(16) __bf16 sB[32 * KCHUNK];

  const unsigned tid  = threadIdx.x;
  const unsigned wave = tid >> 5;
  const unsigned lane = tid & 31;
  const unsigned lr   = lane & 15;     // row (A) / column (B) within tile
  const unsigned half = lane >> 4;     // K-half selector per ISA fragment layout
  const unsigned nt   = blockIdx.x * 32;
  const unsigned mt   = wave * 32;

  v8f acc00 = {0.f,0.f,0.f,0.f,0.f,0.f,0.f,0.f};  // (mt,    nt)
  v8f acc10 = acc00;                               // (mt+16, nt)
  v8f acc01 = acc00;                               // (mt,    nt+16)
  v8f acc11 = acc00;                               // (mt+16, nt+16)

  // Cooperative-load assignment: 8 threads per stripe row, 16B chunks.
  const unsigned ldrow = tid >> 3;          // 0..31
  const unsigned ldco0 = (tid & 7) * 8;     // starting half-offset within row

  for (int k0 = 0; k0 < K; k0 += KCHUNK) {
    const int kc = (K - k0 < KCHUNK) ? (K - k0) : KCHUNK;

    // Async-stage W rows nt..nt+31, cols k0..k0+kc into LDS (row-major [32][kc]).
    // LDS destination offset = low 32 bits of the shared-memory address.
    {
      const __bf16* wrow = &W[(unsigned)(nt + ldrow) * (unsigned)K + (unsigned)k0];
      const unsigned lbase = (unsigned)(size_t)(&sB[ldrow * kc]);
      for (int co = (int)ldco0; co < kc; co += 64) {
        const unsigned loff = lbase + 2u * (unsigned)co;
        const __bf16* gp = wrow + co;
        asm volatile("global_load_async_to_lds_b128 %0, %1, off"
                     :: "v"(loff), "v"(gp) : "memory");
      }
      asm volatile("s_wait_asynccnt 0x0" ::: "memory");
    }
    __syncthreads();

    for (int kk = 0; kk < kc; kk += 32) {
      // B fragments: columns (nt+lr) and (nt+16+lr) of W^T, from LDS.
      const __bf16* bp0 = &sB[lr * kc + kk + half * 8];
      const __bf16* bp1 = bp0 + 16u * (unsigned)kc;
      const v16bf bf0 = cat16(*(const v8bf*)bp0, *(const v8bf*)(bp0 + 16));
      const v16bf bf1 = cat16(*(const v8bf*)bp1, *(const v8bf*)(bp1 + 16));

      // A fragments: rows mt..mt+15 and mt+16..mt+31, from global (hot in L2).
      const __bf16* ap0 = &A[(mt + lr) * (unsigned)K + (unsigned)(k0 + kk) + half * 8];
      __builtin_prefetch(ap0 + 32, 0, 3);   // global_prefetch_b8: next K slice
      const __bf16* ap1 = ap0 + 16u * (unsigned)K;
      const v16bf af0 = cat16(*(const v8bf*)ap0, *(const v8bf*)(ap0 + 16));
      const v16bf af1 = cat16(*(const v8bf*)ap1, *(const v8bf*)(ap1 + 16));

      acc00 = __builtin_amdgcn_wmma_f32_16x16x32_bf16(
                  false, af0, false, bf0, (short)0, acc00, false, false);
      acc10 = __builtin_amdgcn_wmma_f32_16x16x32_bf16(
                  false, af1, false, bf0, (short)0, acc10, false, false);
      acc01 = __builtin_amdgcn_wmma_f32_16x16x32_bf16(
                  false, af0, false, bf1, (short)0, acc01, false, false);
      acc11 = __builtin_amdgcn_wmma_f32_16x16x32_bf16(
                  false, af1, false, bf1, (short)0, acc11, false, false);
    }
    __syncthreads();
  }

  const unsigned col0 = nt + lr;
  const unsigned col1 = nt + 16 + lr;
  const unsigned mr0  = mt + 8 * half;
  const unsigned mr1  = mt + 16 + 8 * half;
  epilogue<MODE>(acc00, mr0, col0, bias, inp, h, heval, ksum, hb_out, out, ldout);
  epilogue<MODE>(acc10, mr1, col0, bias, inp, h, heval, ksum, hb_out, out, ldout);
  epilogue<MODE>(acc01, mr0, col1, bias, inp, h, heval, ksum, hb_out, out, ldout);
  epilogue<MODE>(acc11, mr1, col1, bias, inp, h, heval, ksum, hb_out, out, ldout);
}

// ---------------------------------------------------------------------------
// Utility kernels
// ---------------------------------------------------------------------------
__global__ void f32_to_bf16_k(const float* __restrict__ src, __bf16* __restrict__ dst, unsigned n) {
  for (unsigned i = blockIdx.x * blockDim.x + threadIdx.x; i < n;
       i += gridDim.x * blockDim.x)
    dst[i] = (__bf16)src[i];
}

__global__ void zero_f32_k(float* __restrict__ p, unsigned n) {
  for (unsigned i = blockIdx.x * blockDim.x + threadIdx.x; i < n;
       i += gridDim.x * blockDim.x)
    p[i] = 0.0f;
}

__global__ void zero_bf16_k(__bf16* __restrict__ p, unsigned n) {
  for (unsigned i = blockIdx.x * blockDim.x + threadIdx.x; i < n;
       i += gridDim.x * blockDim.x)
    p[i] = (__bf16)0.0f;
}

// xb[b][i] = bf16( x[b][t][i] )
__global__ void slice_x_k(const float* __restrict__ x, __bf16* __restrict__ xb, int t) {
  for (unsigned idx = blockIdx.x * blockDim.x + threadIdx.x; idx < (unsigned)(B_ * I_);
       idx += gridDim.x * blockDim.x) {
    const unsigned b = idx >> 8;            // I_ == 256
    const unsigned i = idx & 255u;
    xb[idx] = (__bf16)x[(b * S_ + (unsigned)t) * I_ + i];
  }
}

// ---------------------------------------------------------------------------
extern "C" void kernel_launch(void* const* d_in, const int* in_sizes, int n_in,
                              void* d_out, int out_size, void* d_ws, size_t ws_size,
                              hipStream_t stream) {
  (void)in_sizes; (void)n_in; (void)out_size; (void)ws_size;
  const float* x     = (const float*)d_in[0];   // (B,S,I)
  const float* W_rec = (const float*)d_in[1];   // (H,H)
  const float* W_in  = (const float*)d_in[2];   // (H,I)
  const float* b_in  = (const float*)d_in[3];   // (H)
  const float* W_dec = (const float*)d_in[4];   // (O,H)
  const float* b_dec = (const float*)d_in[5];   // (O)
  float* out = (float*)d_out;                   // (B,O)

  // Workspace carve-up (256-byte aligned)
  char* w = (char*)d_ws;
  auto carve = [&](size_t bytes) { void* p = (void*)w; w += (bytes + 255) & ~(size_t)255; return p; };
  __bf16* Wrec_b = (__bf16*)carve((size_t)H_ * H_ * 2);
  __bf16* Win_b  = (__bf16*)carve((size_t)H_ * I_ * 2);
  __bf16* Wdec_b = (__bf16*)carve((size_t)O_ * H_ * 2);
  float*  h      = (float*) carve((size_t)B_ * H_ * 4);
  float*  heval  = (float*) carve((size_t)B_ * H_ * 4);
  float*  ksum   = (float*) carve((size_t)B_ * H_ * 4);
  float*  inp    = (float*) carve((size_t)B_ * H_ * 4);
  __bf16* hbA    = (__bf16*)carve((size_t)B_ * H_ * 2);  // double-buffered bf16 h_eval
  __bf16* hbB    = (__bf16*)carve((size_t)B_ * H_ * 2);
  __bf16* xb     = (__bf16*)carve((size_t)B_ * I_ * 2);

  // One-time per launch: weight conversion + state init (deterministic).
  f32_to_bf16_k<<<2048, 256, 0, stream>>>(W_rec, Wrec_b, (unsigned)(H_ * H_));
  f32_to_bf16_k<<<1024, 256, 0, stream>>>(W_in,  Win_b,  (unsigned)(H_ * I_));
  f32_to_bf16_k<<<1024, 256, 0, stream>>>(W_dec, Wdec_b, (unsigned)(O_ * H_));
  zero_f32_k <<<512, 256, 0, stream>>>(h,     (unsigned)(B_ * H_));
  zero_f32_k <<<512, 256, 0, stream>>>(heval, (unsigned)(B_ * H_));
  zero_bf16_k<<<512, 256, 0, stream>>>(hbA,   (unsigned)(B_ * H_));

  const dim3 blk(256);
  const dim3 gH(H_ / 32);   // 64 N-stripes for H-wide outputs
  const dim3 gO(O_ / 32);   // 8 stripes for decode

  for (int t = 0; t < S_; ++t) {
    // inp = x_t @ W_in^T + b_in  (shared across the 4 RK4 evals)
    slice_x_k<<<256, 256, 0, stream>>>(x, xb, t);
    rk4_gemm<0><<<gH, blk, 0, stream>>>(xb, Win_b, I_, b_in, nullptr,
                                        nullptr, nullptr, nullptr, nullptr, inp, H_);
    // RK4 stages: each reads the previous stage's bf16 h_eval, writes the next
    rk4_gemm<1><<<gH, blk, 0, stream>>>(hbA, Wrec_b, H_, nullptr, inp,
                                        h, heval, ksum, hbB, nullptr, H_);
    rk4_gemm<2><<<gH, blk, 0, stream>>>(hbB, Wrec_b, H_, nullptr, inp,
                                        h, heval, ksum, hbA, nullptr, H_);
    rk4_gemm<3><<<gH, blk, 0, stream>>>(hbA, Wrec_b, H_, nullptr, inp,
                                        h, heval, ksum, hbB, nullptr, H_);
    rk4_gemm<4><<<gH, blk, 0, stream>>>(hbB, Wrec_b, H_, nullptr, inp,
                                        h, heval, ksum, hbA, nullptr, H_);
  }

  // Decode: out = h_final @ W_dec^T + b_dec  (hbA holds bf16(h_final))
  rk4_gemm<5><<<gO, blk, 0, stream>>>(hbA, Wdec_b, H_, b_dec, nullptr,
                                      nullptr, nullptr, nullptr, nullptr, out, O_);
}
